// PointNet2SemanticSegmentation_10625749090985
// MI455X (gfx1250) — compile-verified, matrix operand order checked by reading, not run on hardware
//
#include <hip/hip_runtime.h>
#include <hip/hip_bf16.h>
#include <math.h>

// ---------------------------------------------------------------------------
// MI455X (gfx1250, wave32) PointNet++ forward.
// Dense math: v_wmma_f32_16x16x32_f16, 16x32 output tile per wave (A fragment
// shared by two WMMAs), fused bias+BN+ReLU epilogue, f16 activations.
// FPS: dmin in registers, xyz staged in LDS for N<=1024 levels.
// ---------------------------------------------------------------------------

typedef __attribute__((ext_vector_type(16))) _Float16 v16h;
typedef __attribute__((ext_vector_type(8)))  _Float16 v8h;
typedef __attribute__((ext_vector_type(8)))  float    v8f;

static __device__ __forceinline__ void load_h8(const _Float16* __restrict__ p,
                                               v16h& f, int base) {
  const v8h v = *reinterpret_cast<const v8h*>(p);
#pragma unroll
  for (int i = 0; i < 8; ++i) f[base + i] = v[i];
}

// ------------------------- farthest point sampling -------------------------
// dmin lives in registers (NPT floats/thread). For small levels xyz is staged
// in LDS; level 0 (N=16384, 196KB) reads xyz from global (L2/WGP$-resident).
template <int TB, int NPT, bool LDSXYZ, int NMAX>
__global__ void __launch_bounds__(TB)
fps_reg_kernel(const float* __restrict__ xyz, int* __restrict__ fidx,
               int Nn, int npoint) {
  const int b = blockIdx.x;
  const float* px = xyz + (size_t)b * Nn * 3;
  int* out = fidx + (size_t)b * npoint;
  __shared__ float sval[TB];
  __shared__ int   sidx[TB];
  __shared__ int   s_far;
  __shared__ float sx[NMAX], sy[NMAX], sz[NMAX];

  float dm[NPT];
#pragma unroll
  for (int j = 0; j < NPT; ++j) dm[j] = 1e10f;
  if (LDSXYZ) {
    for (int i = threadIdx.x; i < Nn; i += TB) {
      sx[i] = px[i * 3 + 0];
      sy[i] = px[i * 3 + 1];
      sz[i] = px[i * 3 + 2];
    }
  }
  if (threadIdx.x == 0) s_far = 0;
  __syncthreads();

  for (int it = 0; it < npoint; ++it) {
    const int far = s_far;
    if (threadIdx.x == 0) out[it] = far;
    float cx, cy, cz;
    if (LDSXYZ) { cx = sx[far]; cy = sy[far]; cz = sz[far]; }
    else        { cx = px[far * 3 + 0]; cy = px[far * 3 + 1]; cz = px[far * 3 + 2]; }
    float best = -1.0f; int besti = 0;
#pragma unroll
    for (int j = 0; j < NPT; ++j) {
      const int i = threadIdx.x + j * TB;
      float X, Y, Z;
      if (LDSXYZ) { X = sx[i]; Y = sy[i]; Z = sz[i]; }
      else        { X = px[i * 3 + 0]; Y = px[i * 3 + 1]; Z = px[i * 3 + 2]; }
      const float dx = X - cx, dy = Y - cy, dz = Z - cz;
      const float d  = dx * dx + dy * dy + dz * dz;
      dm[j] = dm[j] < d ? dm[j] : d;
      if (dm[j] > best) { best = dm[j]; besti = i; }
    }
    sval[threadIdx.x] = best; sidx[threadIdx.x] = besti;
    __syncthreads();
#pragma unroll
    for (int off = TB / 2; off > 0; off >>= 1) {
      if ((int)threadIdx.x < off) {
        const float v2 = sval[threadIdx.x + off];
        const int   i2 = sidx[threadIdx.x + off];
        if (v2 > sval[threadIdx.x] ||
            (v2 == sval[threadIdx.x] && i2 < sidx[threadIdx.x])) {
          sval[threadIdx.x] = v2; sidx[threadIdx.x] = i2;
        }
      }
      __syncthreads();
    }
    if (threadIdx.x == 0) s_far = sidx[0];
    __syncthreads();
  }
}

__global__ void gather_xyz_kernel(const float* __restrict__ xyz, const int* __restrict__ fidx,
                                  float* __restrict__ out, int Nn, int S, int total) {
  const int t = blockIdx.x * blockDim.x + threadIdx.x;
  if (t >= total) return;
  const int b = t / S;
  const int i = fidx[t];
  const float* p = xyz + ((size_t)b * Nn + i) * 3;
  float* o = out + (size_t)t * 3;
  o[0] = p[0]; o[1] = p[1]; o[2] = p[2];
}

// ------------------------------ ball query ---------------------------------
// One wave per query. Ascending-order scan + ballot compaction reproduces the
// reference sort-then-truncate semantics; leftover slots <- first hit.
__global__ void ball_query_kernel(const float* __restrict__ xyz,
                                  const float* __restrict__ new_xyz,
                                  int* __restrict__ gidx,
                                  int Nn, int S, int ns, float r2) {
  const int q = blockIdx.x;           // b*S + s
  const int b = q / S;
  const float* px = xyz + (size_t)b * Nn * 3;
  const float* c  = new_xyz + (size_t)q * 3;
  int* out = gidx + (size_t)q * ns;
  const float cx = c[0], cy = c[1], cz = c[2];
  const int lane = threadIdx.x;
  int count = 0;
  int firstIdx = 0;
  bool haveFirst = false;
  for (int base = 0; base < Nn && count < ns; base += 32) {
    const int i = base + lane;
    const float dx = px[i * 3 + 0] - cx;
    const float dy = px[i * 3 + 1] - cy;
    const float dz = px[i * 3 + 2] - cz;
    const bool in = (dx * dx + dy * dy + dz * dz) <= r2;
    const unsigned mask = (unsigned)__ballot(in);
    if (!haveFirst && mask != 0u) { firstIdx = base + (__ffs(mask) - 1); haveFirst = true; }
    if (in) {
      const int pos = count + __popc(mask & ((1u << lane) - 1u));
      if (pos < ns) out[pos] = i;
    }
    count += __popc(mask);
  }
  const int total = count < ns ? count : ns;
  for (int p = total + lane; p < ns; p += 32) out[p] = firstIdx;
}

// ------------------------ grouping (SA feature build) -----------------------
__global__ void sa_group_kernel(const float* __restrict__ xyz,
                                const float* __restrict__ new_xyz,
                                const _Float16* __restrict__ pts_h,
                                const float* __restrict__ pts_f,
                                const int* __restrict__ gidx,
                                _Float16* __restrict__ out,
                                int Nn, int S, int ns, int Cpts, int Kpad, int total) {
  const int t = blockIdx.x * blockDim.x + threadIdx.x;
  if (t >= total) return;
  const int b = t / (S * ns);
  const int rem = t - b * S * ns;
  const int s = rem / ns;
  const int i = gidx[t];
  const float* p = xyz + ((size_t)b * Nn + i) * 3;
  const float* c = new_xyz + ((size_t)b * S + s) * 3;
  _Float16* o = out + (size_t)t * Kpad;
  o[0] = (_Float16)(p[0] - c[0]);
  o[1] = (_Float16)(p[1] - c[1]);
  o[2] = (_Float16)(p[2] - c[2]);
  if (pts_f) {
    const float* q = pts_f + ((size_t)b * Nn + i) * Cpts;
    for (int j = 0; j < Cpts; ++j) o[3 + j] = (_Float16)q[j];
  } else {
    const _Float16* q = pts_h + ((size_t)b * Nn + i) * Cpts;
    for (int j = 0; j < Cpts; ++j) o[3 + j] = q[j];
  }
  for (int j = 3 + Cpts; j < Kpad; ++j) o[j] = (_Float16)0.f;
}

// ------------------------------ WMMA GEMM -----------------------------------
// 16x32 output tile per wave: one A fragment feeds two v_wmma on adjacent
// B tiles -> halves A traffic, doubles WMMA density per K iteration.
// A: [M x Kpad] f16 row-major; Wt: [Cpad32 x Kpad] (row c = output column c).
__global__ void __launch_bounds__(256)
gemm_wmma_kernel(const _Float16* __restrict__ A,
                 const _Float16* __restrict__ Wt,
                 const float* __restrict__ fscale,
                 const float* __restrict__ fbias,
                 _Float16* __restrict__ Oh,
                 float* __restrict__ Of,
                 int M, int Kpad, int C, int ldo, int relu) {
  const int lane = threadIdx.x & 31;
  const int wave = threadIdx.x >> 5;
  const int tilesN2 = (C + 31) >> 5;          // pairs of 16-wide column tiles
  const int tilesM  = M >> 4;
  const int tileId  = blockIdx.x * 8 + wave;
  if (tileId >= tilesM * tilesN2) return;     // wave-uniform: EXEC stays all-1s
  const int tm = tileId / tilesN2;
  const int tn = (tileId - tm * tilesN2) * 2;

  const int half = lane >> 4;                 // lane group -> K sub-chunks
  const int l16  = lane & 15;
  const int ka = half * 8;                    // K = {0..7} or {8..15}
  const int kb = 16 + half * 8;               // K = {16..23} or {24..31}

  const _Float16* __restrict__ arow  = A  + (size_t)(tm * 16 + l16) * Kpad;
  const _Float16* __restrict__ brow0 = Wt + (size_t)(tn * 16 + l16) * Kpad;
  const _Float16* __restrict__ brow1 = brow0 + (size_t)16 * Kpad;

  v8f acc0 = {0.f, 0.f, 0.f, 0.f, 0.f, 0.f, 0.f, 0.f};
  v8f acc1 = {0.f, 0.f, 0.f, 0.f, 0.f, 0.f, 0.f, 0.f};
  for (int k = 0; k < Kpad; k += 32) {
    v16h af, bf0, bf1;
    load_h8(arow  + k + ka, af,  0);
    load_h8(arow  + k + kb, af,  8);
    load_h8(brow0 + k + ka, bf0, 0);
    load_h8(brow0 + k + kb, bf0, 8);
    load_h8(brow1 + k + ka, bf1, 0);
    load_h8(brow1 + k + kb, bf1, 8);
    if (k + 32 < Kpad) {
      __builtin_prefetch(arow + k + 32, 0, 1);   // global_prefetch_b8
    }
    acc0 = __builtin_amdgcn_wmma_f32_16x16x32_f16(false, af, false, bf0,
                                                  (short)0, acc0, false, false);
    acc1 = __builtin_amdgcn_wmma_f32_16x16x32_f16(false, af, false, bf1,
                                                  (short)0, acc1, false, false);
  }
#pragma unroll
  for (int r = 0; r < 8; ++r) {
    const int row  = tm * 16 + r + half * 8;   // C/D layout: VGPR r -> M=r / r+8
    const int col0 = tn * 16 + l16;
    const int col1 = col0 + 16;
    if (col0 < C) {
      float v = acc0[r] * fscale[col0] + fbias[col0];
      if (relu) v = fmaxf(v, 0.f);
      if (Oh) Oh[(size_t)row * ldo + col0] = (_Float16)v;
      else    Of[(size_t)row * ldo + col0] = v;
    }
    if (col1 < C) {
      float v = acc1[r] * fscale[col1] + fbias[col1];
      if (relu) v = fmaxf(v, 0.f);
      if (Oh) Oh[(size_t)row * ldo + col1] = (_Float16)v;
      else    Of[(size_t)row * ldo + col1] = v;
    }
  }
}

// ------------------------------- max pool -----------------------------------
__global__ void max_pool_kernel(const _Float16* __restrict__ feat,
                                _Float16* __restrict__ pooled,
                                int rows, int ns, int C, int ldf, int total) {
  const int t = blockIdx.x * blockDim.x + threadIdx.x;
  if (t >= total) return;
  const int r = t / C, c = t - r * C;
  const _Float16* p = feat + (size_t)r * ns * ldf + c;
  float m = -3.0e38f;
  for (int k = 0; k < ns; ++k) {
    const float v = (float)p[(size_t)k * ldf];
    m = v > m ? v : m;
  }
  pooled[t] = (_Float16)m;
}

// -------------------- feature propagation: 3-NN interp ----------------------
__global__ void fp_interp_kernel(const float* __restrict__ xyz1,
                                 const float* __restrict__ xyz2,
                                 const _Float16* __restrict__ pts1,
                                 const _Float16* __restrict__ pts2,
                                 _Float16* __restrict__ out,
                                 int n1, int n2, int C1, int C2, int total) {
  const int t = blockIdx.x * blockDim.x + threadIdx.x;
  if (t >= total) return;
  const int b = t / n1;
  const float* p1  = xyz1 + (size_t)t * 3;
  const float* p2b = xyz2 + (size_t)b * n2 * 3;
  const float x = p1[0], y = p1[1], z = p1[2];
  float d0 = 3.4e38f, d1 = 3.4e38f, d2 = 3.4e38f;
  int   i0 = 0, i1 = 0, i2 = 0;
  for (int j = 0; j < n2; ++j) {
    const float dx = p2b[j * 3 + 0] - x;
    const float dy = p2b[j * 3 + 1] - y;
    const float dz = p2b[j * 3 + 2] - z;
    const float d = dx * dx + dy * dy + dz * dz;
    if (d < d0)      { d2 = d1; i2 = i1; d1 = d0; i1 = i0; d0 = d; i0 = j; }
    else if (d < d1) { d2 = d1; i2 = i1; d1 = d;  i1 = j; }
    else if (d < d2) { d2 = d;  i2 = j; }
  }
  float r0 = 1.f / (d0 + 1e-8f), r1 = 1.f / (d1 + 1e-8f), r2 = 1.f / (d2 + 1e-8f);
  const float rs = r0 + r1 + r2;
  r0 /= rs; r1 /= rs; r2 /= rs;
  const int Kpad = C1 + C2;   // always a multiple of 32 here
  _Float16* o = out + (size_t)t * Kpad;
  if (pts1) {
    const _Float16* q = pts1 + (size_t)t * C1;
    for (int c = 0; c < C1; ++c) o[c] = q[c];
  }
  const _Float16* q0 = pts2 + ((size_t)b * n2 + i0) * C2;
  const _Float16* q1 = pts2 + ((size_t)b * n2 + i1) * C2;
  const _Float16* q2 = pts2 + ((size_t)b * n2 + i2) * C2;
  for (int c = 0; c < C2; ++c)
    o[C1 + c] = (_Float16)(r0 * (float)q0[c] + r1 * (float)q1[c] + r2 * (float)q2[c]);
}

// ------------------------------ log softmax ---------------------------------
__global__ void log_softmax_kernel(const float* __restrict__ logits,
                                   float* __restrict__ out, int total, int C) {
  const int r = blockIdx.x * blockDim.x + threadIdx.x;
  if (r >= total) return;
  const float* p = logits + (size_t)r * C;
  float m = p[0];
  for (int c = 1; c < C; ++c) m = fmaxf(m, p[c]);
  float s = 0.f;
  for (int c = 0; c < C; ++c) s += __expf(p[c] - m);
  const float ls = __logf(s);
  float* o = out + (size_t)r * C;
  for (int c = 0; c < C; ++c) o[c] = p[c] - m - ls;
}

// ------------------------- weight pack + BN fold ----------------------------
__global__ void pack_layer_kernel(const float* __restrict__ W, const float* __restrict__ b,
                                  const float* __restrict__ g, const float* __restrict__ bt,
                                  _Float16* __restrict__ Wt,
                                  float* __restrict__ fs, float* __restrict__ fb,
                                  int K, int C, int Kpad, int Cpad) {
  const int t = blockIdx.x * blockDim.x + threadIdx.x;
  const int totalW = Cpad * Kpad;
  if (t < totalW) {
    const int c = t / Kpad, k = t - c * Kpad;
    const float v = (k < K && c < C) ? W[(size_t)k * C + c] : 0.f;
    Wt[t] = (_Float16)v;
  }
  if (t < Cpad) {
    float s = 0.f, sh = 0.f;
    if (t < C) {
      if (g) {
        const float inv = rsqrtf(1.f + 1e-5f);
        s = g[t] * inv;
        sh = b[t] * s + bt[t];
      } else {
        s = 1.f;
        sh = b[t];
      }
    }
    fs[t] = s; fb[t] = sh;
  }
}

// ---------------------------------------------------------------------------
// Host orchestration
// ---------------------------------------------------------------------------
static inline int pad32(int x) { return (x + 31) & ~31; }

static void run_gemm(hipStream_t stream, const _Float16* A, const _Float16* Wt,
                     const float* fs, const float* fb,
                     _Float16* Oh, float* Of,
                     int M, int Kpad, int C, int ldo, int relu) {
  const int tiles = (M >> 4) * ((C + 31) >> 5);
  const int blocks = (tiles + 7) / 8;
  gemm_wmma_kernel<<<blocks, 256, 0, stream>>>(A, Wt, fs, fb, Oh, Of, M, Kpad, C, ldo, relu);
}

extern "C" void kernel_launch(void* const* d_in, const int* in_sizes, int n_in,
                              void* d_out, int out_size, void* d_ws, size_t ws_size,
                              hipStream_t stream) {
  const int B = 4, N = 16384, NS = 32, NCLS = 20;

  // ---- detect pytree flattening order (xyz size == fp[0].W size == 196608) ----
  int pstart = 0;
  if (n_in > 1 && in_sizes[0] == B * N * 3 && in_sizes[1] != 256) pstart = 1;
  const int xyz_i = pstart ? 0 : (n_in - 1);
  const bool sa_first = (in_sizes[pstart] == 6 * 32);   // sa0 L0 W = 192

  const int fp_nl[4] = {2, 2, 2, 3};
  int sa_i[4][3], fp_i[4][3], h1_i, h2w_i, h2b_i;
  {
    int p = pstart;
    if (sa_first) {
      for (int s = 0; s < 4; ++s) for (int l = 0; l < 3; ++l) { sa_i[s][l] = p; p += 4; }
      for (int f = 0; f < 4; ++f) for (int l = 0; l < fp_nl[f]; ++l) { fp_i[f][l] = p; p += 4; }
      h1_i = p; p += 4; h2w_i = p++; h2b_i = p++;
    } else {
      for (int f = 0; f < 4; ++f) for (int l = 0; l < fp_nl[f]; ++l) { fp_i[f][l] = p; p += 4; }
      h1_i = p; p += 4; h2w_i = p++; h2b_i = p++;
      for (int s = 0; s < 4; ++s) for (int l = 0; l < 3; ++l) { sa_i[s][l] = p; p += 4; }
    }
  }
  const float* xyz = (const float*)d_in[xyz_i];

  // ---- workspace bump allocator ----
  char* wsbase = (char*)d_ws;
  size_t off = 0;
  auto alloc = [&](size_t bytes) -> void* {
    off = (off + 255) & ~(size_t)255;
    void* p = wsbase + off;
    off += bytes;
    return p;
  };

  // ---- layer table: 12 SA + 9 FP + head1 + head2 ----
  struct LayerW {
    const float *W, *b, *g, *bt;
    int K, C, Kpad, Cpad;
    _Float16* Wt; float* fs; float* fb;
  };
  LayerW L[23];
  int nL = 0;
  const int sa_ch[4][4] = {{6,32,32,64},{67,64,64,128},{131,128,128,256},{259,256,256,512}};
  const int fp_ch[4][4] = {{768,256,256,0},{384,256,256,0},{320,256,128,0},{128,128,128,128}};
  for (int s = 0; s < 4; ++s)
    for (int l = 0; l < 3; ++l) {
      const int li = sa_i[s][l];
      L[nL++] = { (const float*)d_in[li], (const float*)d_in[li+1],
                  (const float*)d_in[li+2], (const float*)d_in[li+3],
                  sa_ch[s][l], sa_ch[s][l+1], 0,0, nullptr,nullptr,nullptr };
    }
  for (int f = 0; f < 4; ++f)
    for (int l = 0; l < fp_nl[f]; ++l) {
      const int li = fp_i[f][l];
      L[nL++] = { (const float*)d_in[li], (const float*)d_in[li+1],
                  (const float*)d_in[li+2], (const float*)d_in[li+3],
                  fp_ch[f][l], fp_ch[f][l+1], 0,0, nullptr,nullptr,nullptr };
    }
  L[nL++] = { (const float*)d_in[h1_i], (const float*)d_in[h1_i+1],
              (const float*)d_in[h1_i+2], (const float*)d_in[h1_i+3],
              128, 128, 0,0, nullptr,nullptr,nullptr };
  L[nL++] = { (const float*)d_in[h2w_i], (const float*)d_in[h2b_i],
              nullptr, nullptr, 128, NCLS, 0,0, nullptr,nullptr,nullptr };

  for (int i = 0; i < 23; ++i) {
    L[i].Kpad = pad32(L[i].K);
    L[i].Cpad = pad32(L[i].C);          // pad to 32 so 16x32 wave tiles are safe
    L[i].Wt = (_Float16*)alloc((size_t)L[i].Cpad * L[i].Kpad * sizeof(_Float16));
    L[i].fs = (float*)alloc((size_t)L[i].Cpad * sizeof(float));
    L[i].fb = (float*)alloc((size_t)L[i].Cpad * sizeof(float));
    const int tot = L[i].Cpad * L[i].Kpad;
    pack_layer_kernel<<<(tot + 255) / 256, 256, 0, stream>>>(
        L[i].W, L[i].b, L[i].g, L[i].bt, L[i].Wt, L[i].fs, L[i].fb,
        L[i].K, L[i].C, L[i].Kpad, L[i].Cpad);
  }

  // ---- activation scratch ----
  int* fidx = (int*)alloc((size_t)B * 1024 * sizeof(int));
  int* gq   = (int*)alloc((size_t)B * 1024 * NS * sizeof(int));
  const int Ssa[4] = {1024, 256, 64, 16};
  const float rad[4] = {0.1f, 0.2f, 0.4f, 0.8f};
  float* newx[4];
  for (int s = 0; s < 4; ++s) newx[s] = (float*)alloc((size_t)B * Ssa[s] * 3 * sizeof(float));
  const size_t BUF_ELEMS = (size_t)131072 * 64;   // max rows*Kpad over all GEMMs
  _Float16* pp[2];
  pp[0] = (_Float16*)alloc(BUF_ELEMS * sizeof(_Float16));
  pp[1] = (_Float16*)alloc(BUF_ELEMS * sizeof(_Float16));
  _Float16* pooled[4];
  const int poolC[4] = {64, 128, 256, 512};
  for (int s = 0; s < 4; ++s)
    pooled[s] = (_Float16*)alloc((size_t)B * Ssa[s] * poolC[s] * sizeof(_Float16));
  _Float16* fpo[4];
  const int fpoN[4] = {64, 256, 1024, 16384};
  const int fpoC[4] = {256, 256, 128, 128};
  for (int f = 0; f < 4; ++f)
    fpo[f] = (_Float16*)alloc((size_t)B * fpoN[f] * fpoC[f] * sizeof(_Float16));
  float* logits = (float*)alloc((size_t)B * N * NCLS * sizeof(float));

  // ---------------------------- SA levels ---------------------------------
  const float* cur_xyz = xyz;
  int curN = N;
  const _Float16* cur_pts_h = nullptr;
  const float* cur_pts_f = xyz;
  int curC = 3;
  for (int s = 0; s < 4; ++s) {
    const int S = Ssa[s];
    switch (s) {  // register-dmin FPS; LDS-resident xyz for small levels
      case 0: fps_reg_kernel<256, 64, false,    1><<<B, 256, 0, stream>>>(cur_xyz, fidx, curN, S); break;
      case 1: fps_reg_kernel<256,  4, true,  1024><<<B, 256, 0, stream>>>(cur_xyz, fidx, curN, S); break;
      case 2: fps_reg_kernel<256,  1, true,   256><<<B, 256, 0, stream>>>(cur_xyz, fidx, curN, S); break;
      default:fps_reg_kernel< 64,  1, true,    64><<<B,  64, 0, stream>>>(cur_xyz, fidx, curN, S); break;
    }
    gather_xyz_kernel<<<(B * S + 255) / 256, 256, 0, stream>>>(
        cur_xyz, fidx, newx[s], curN, S, B * S);
    ball_query_kernel<<<B * S, 32, 0, stream>>>(
        cur_xyz, newx[s], gq, curN, S, NS, rad[s] * rad[s]);
    const int rows = B * S * NS;
    const int Kpad0 = pad32(3 + curC);
    sa_group_kernel<<<(rows + 255) / 256, 256, 0, stream>>>(
        cur_xyz, newx[s], cur_pts_h, cur_pts_f, gq, pp[0],
        curN, S, NS, curC, Kpad0, rows);
    int cur = 0;
    for (int l = 0; l < 3; ++l) {
      const LayerW& lw = L[s * 3 + l];
      run_gemm(stream, pp[cur], lw.Wt, lw.fs, lw.fb, pp[1 - cur], nullptr,
               rows, lw.Kpad, lw.C, lw.C, 1);
      cur ^= 1;
    }
    const int Cout = sa_ch[s][3];
    max_pool_kernel<<<(B * S * Cout + 255) / 256, 256, 0, stream>>>(
        pp[cur], pooled[s], B * S, NS, Cout, Cout, B * S * Cout);
    cur_xyz = newx[s]; curN = S;
    cur_pts_h = pooled[s]; cur_pts_f = nullptr; curC = Cout;
  }

  // ---------------------------- FP levels ---------------------------------
  struct FPc {
    const float *x1, *x2; int n1, n2;
    const _Float16* p1; int C1; const _Float16* p2; int C2;
    _Float16* out; int lbase; int nl;
  };
  const FPc fpc[4] = {
    { newx[2], newx[3],   64,   16, pooled[2], 256, pooled[3], 512, fpo[0], 12, 2 },
    { newx[1], newx[2],  256,   64, pooled[1], 128, fpo[0],    256, fpo[1], 14, 2 },
    { newx[0], newx[1], 1024,  256, pooled[0],  64, fpo[1],    256, fpo[2], 16, 2 },
    { xyz,     newx[0],    N, 1024, nullptr,     0, fpo[2],    128, fpo[3], 18, 3 },
  };
  for (int f = 0; f < 4; ++f) {
    const FPc& c = fpc[f];
    const int rows = B * c.n1;
    fp_interp_kernel<<<(rows + 255) / 256, 256, 0, stream>>>(
        c.x1, c.x2, c.p1, c.p2, pp[0], c.n1, c.n2, c.C1, c.C2, rows);
    int cur = 0;
    for (int l = 0; l < c.nl; ++l) {
      const LayerW& lw = L[c.lbase + l];
      _Float16* dst = (l == c.nl - 1) ? c.out : pp[1 - cur];
      run_gemm(stream, pp[cur], lw.Wt, lw.fs, lw.fb, dst, nullptr,
               rows, lw.Kpad, lw.C, lw.C, 1);
      cur ^= 1;
    }
  }

  // ------------------------------ head ------------------------------------
  const int Mh = B * N;
  run_gemm(stream, fpo[3], L[21].Wt, L[21].fs, L[21].fb, pp[0], nullptr,
           Mh, 128, 128, 128, 1);
  run_gemm(stream, pp[0], L[22].Wt, L[22].fs, L[22].fb, nullptr, logits,
           Mh, 128, NCLS, NCLS, 0);
  log_softmax_kernel<<<(Mh + 255) / 256, 256, 0, stream>>>(
      logits, (float*)d_out, Mh, NCLS);
  (void)out_size; (void)ws_size;
}
